// ScannerChunker_58445914964629
// MI455X (gfx1250) — compile-verified, matrix operand order checked by reading, not run on hardware
//
#include <hip/hip_runtime.h>
#include <hip/hip_bf16.h>

// ---- problem constants (from reference) ----
#define BB 2048
#define VV 256
#define TT 11
#define KK 8
#define NN 8
#define HH 255          // combination neurons per slot
#define KH (KK * HH)    // 2040
#define ROWS (BB * TT)  // 22528 rows of (b,t)
#define MTILES (ROWS / 16)  // 1408 exact
#define HTILES 16       // 255 padded to 256 -> 16 tiles of 16

typedef __attribute__((ext_vector_type(16))) _Float16 v16h;
typedef __attribute__((ext_vector_type(8)))  _Float16 v8h;
typedef __attribute__((ext_vector_type(8)))  float    v8f;
typedef __attribute__((ext_vector_type(4)))  float    v4f;   // native vec for NT loads

// =====================================================================
// Kernel 1: pack hidden_w (K*H, N) f32 -> WMMA B-operand f16 tiles.
// B is the 32x16 f16 operand of v_wmma_f32_16x16x32_f16:
//   lane L (0..15) holds col N=L; element e (0..7) holds K=e.
//   lanes 16..31 hold K=8..15 and elements 8..15 hold K>=16 -> all zero
//   (we only have K=8 real reduction terms = the N scanner outputs).
// Layout: bpack[((kg*16 + ht)*32 + lane)*16 + elem]
// =====================================================================
__global__ void sc_pack_b(const float* __restrict__ hw, _Float16* __restrict__ bpack) {
    int idx = blockIdx.x * blockDim.x + threadIdx.x;   // 0 .. 65535
    if (idx >= 8 * 16 * 32 * 16) return;
    int elem = idx & 15;
    int lane = (idx >> 4) & 31;
    int tile = idx >> 9;          // kg*16 + ht
    int ht   = tile & 15;
    int kg   = tile >> 4;
    float val = 0.0f;
    if (lane < 16 && elem < 8) {
        int h = ht * 16 + lane;   // output-neuron col within the group
        if (h < HH) {
            // hidden_w row-major (K*H, N): [kg*255 + h][n], n == elem
            val = hw[(kg * HH + h) * NN + elem];
        }
    }
    bpack[idx] = (_Float16)val;
}

// =====================================================================
// Kernel 2: stream the 184.5 MB one-hot tensor once (b128, coalesced,
// NON-TEMPORAL so it doesn't evict the L2-resident working set), and for
// each nonzero gather the 8 scanner weights, ReLU, store a packed f16
// row fragment (one 16-byte store).
//   x flat index: (((b*V + v)*T + t)*K + k)
//   h_buf[(b*T + t)*64 + k*8 + n] (f16)  -- matches WMMA A layout rows.
// =====================================================================
__global__ void sc_scan_gather(const float* __restrict__ x,
                               const float* __restrict__ sw,
                               _Float16* __restrict__ hbuf) {
    const int total4 = (BB * VV * TT * KK) / 4;   // 11,534,336
    int i = blockIdx.x * blockDim.x + threadIdx.x;
    const int stride = gridDim.x * blockDim.x;
    const v4f* x4 = reinterpret_cast<const v4f*>(x);
    for (; i < total4; i += stride) {
        v4f xv = __builtin_nontemporal_load(x4 + i);   // read-once stream
#pragma unroll
        for (int c = 0; c < 4; ++c) {
            if (xv[c] != 0.0f) {                 // rare: 1 in 256 elements
                int idx = i * 4 + c;
                int k  = idx & 7;
                int r1 = idx >> 3;               // (b*V + v)*T + t
                int t  = r1 % TT;
                int r2 = r1 / TT;                // b*V + v
                int v  = r2 & (VV - 1);
                int b  = r2 >> 8;
                int row = b * TT + t;
                float xs = xv[c];
                v8h hv;
#pragma unroll
                for (int n = 0; n < NN; ++n) {
                    // scanner_w (N,V,K): [n][v][k]
                    float wv = xs * sw[(n * VV + v) * KK + k];
                    hv[n] = (_Float16)(wv > 0.0f ? wv : 0.0f);
                }
                *reinterpret_cast<v8h*>(hbuf + row * 64 + k * 8) = hv;
            }
        }
    }
}

// =====================================================================
// Kernel 3: initialize out (B,2,T) with the bias.
// =====================================================================
__global__ void sc_init_out(float* __restrict__ out, const float* __restrict__ outb) {
    int i = blockIdx.x * blockDim.x + threadIdx.x;
    if (i < BB * 2 * TT) {
        int o = (i / TT) & 1;
        out[i] = outb[o];
    }
}

// =====================================================================
// Kernel 4: fused grouped GEMM (WMMA) + ReLU + output projection.
// Block = 256 threads = 8 waves. blockIdx.x = M-tile (16 rows of (b,t)).
// wave w handles k-group w. For each of 16 H-tiles:
//   D = wmma_f32_16x16x32_f16(A(h rows), B(hidden_w tile), 0)
//   acc{0,1}[r] += relu(D[r]) * out_w[{0,1}][kg*255 + ht*16 + col]
// Then 16-lane butterfly reduction and atomicAdd into out.
// D layout: lanes 0-15: col=lane, row=r; lanes 16-31: col=lane-16, row=8+r.
// The H-tile loop is kept ROLLED (#pragma unroll 1): full unroll made the
// compiler keep 16 B tiles live and spill to scratch.
// =====================================================================
__global__ void sc_wmma_groupconv(const _Float16* __restrict__ hbuf,
                                  const _Float16* __restrict__ bpack,
                                  const float* __restrict__ outw,
                                  float* __restrict__ out) {
    const int mtile = blockIdx.x;                 // 0 .. 1407
    const int wid   = threadIdx.x >> 5;           // k-group 0..7
    const int lane  = threadIdx.x & 31;
    const int kg    = wid;

    // ---- A operand: 16x32 f16, rows = 16 (b,t) rows, K = n (8 real + 24 zero)
    v16h a;
#pragma unroll
    for (int e = 0; e < 16; ++e) a[e] = (_Float16)0.0f;
    if (lane < 16) {
        v8h frag = *reinterpret_cast<const v8h*>(
            hbuf + (mtile * 16 + lane) * 64 + kg * 8);
#pragma unroll
        for (int e = 0; e < 8; ++e) a[e] = frag[e];
    }

    float acc0[8], acc1[8];
#pragma unroll
    for (int r = 0; r < 8; ++r) { acc0[r] = 0.0f; acc1[r] = 0.0f; }

    const int col = lane & 15;

#pragma unroll 1   // keep rolled: prevents 16 live B tiles + scratch spills
    for (int ht = 0; ht < HTILES; ++ht) {
        // ---- B operand tile (32 bytes/lane, two b128 loads)
        union { v16h v; v4f f4[2]; } bu;
        const v4f* bp = reinterpret_cast<const v4f*>(
            bpack + ((kg * 16 + ht) * 32 + lane) * 16);
        bu.f4[0] = bp[0];
        bu.f4[1] = bp[1];

        // ---- per-lane output-projection weights for this column
        int h = ht * 16 + col;
        float w0 = 0.0f, w1 = 0.0f;
        if (h < HH) {
            int cidx = kg * HH + h;
            w0 = outw[cidx];          // out_w[0][c]
            w1 = outw[KH + cidx];     // out_w[1][c]
        }

        v8f czero = {};
        v8f d = __builtin_amdgcn_wmma_f32_16x16x32_f16(
            /*neg_a=*/false, a, /*neg_b=*/false, bu.v,
            /*c_mod=*/(short)0, czero, /*reuse_a=*/false, /*reuse_b=*/false);

#pragma unroll
        for (int r = 0; r < 8; ++r) {
            float hv = d[r] > 0.0f ? d[r] : 0.0f;   // ReLU on h2
            acc0[r] += hv * w0;
            acc1[r] += hv * w1;
        }
    }

    // ---- reduce across the 16 columns (stays within each half-wave)
#pragma unroll
    for (int off = 1; off < 16; off <<= 1) {
#pragma unroll
        for (int r = 0; r < 8; ++r) {
            acc0[r] += __shfl_xor(acc0[r], off, 32);
            acc1[r] += __shfl_xor(acc1[r], off, 32);
        }
    }

    if ((lane & 15) == 0) {
        int half = lane >> 4;                  // 0 -> rows 0..7, 1 -> rows 8..15
#pragma unroll
        for (int r = 0; r < 8; ++r) {
            int m = r + 8 * half;
            int g = mtile * 16 + m;            // global (b,t) row
            int b = g / TT;
            int t = g % TT;
            atomicAdd(out + (b * 2 + 0) * TT + t, acc0[r]);
            atomicAdd(out + (b * 2 + 1) * TT + t, acc1[r]);
        }
    }
}

// =====================================================================
// launch
// =====================================================================
extern "C" void kernel_launch(void* const* d_in, const int* in_sizes, int n_in,
                              void* d_out, int out_size, void* d_ws, size_t ws_size,
                              hipStream_t stream) {
    const float* x_unfolded = (const float*)d_in[0];   // (B,V,T,K)
    const float* scanner_w  = (const float*)d_in[1];   // (N,V,K)
    const float* hidden_w   = (const float*)d_in[2];   // (K*H, N)
    const float* out_w      = (const float*)d_in[3];   // (2, K*H)
    const float* out_b      = (const float*)d_in[4];   // (2,)
    float* out = (float*)d_out;                        // (B,2,T)

    // workspace layout
    _Float16* hbuf  = (_Float16*)d_ws;                       // ROWS*64 f16 = 2.88 MB
    _Float16* bpack = hbuf + (size_t)ROWS * 64;              // 65536 f16  = 128 KB

    // 1) pack hidden_w into WMMA-B f16 tiles
    sc_pack_b<<<(8 * 16 * 32 * 16) / 256, 256, 0, stream>>>(hidden_w, bpack);

    // 2) stream the one-hot tensor -> packed f16 h rows (bandwidth-bound pass)
    sc_scan_gather<<<4096, 256, 0, stream>>>(x_unfolded, scanner_w, hbuf);

    // 3) bias-init output
    sc_init_out<<<(BB * 2 * TT + 255) / 256, 256, 0, stream>>>(out, out_b);

    // 4) fused WMMA grouped GEMM + ReLU + projection
    sc_wmma_groupconv<<<MTILES, 256, 0, stream>>>(hbuf, bpack, out_w, out);
}
